// LocalDecoderV1_55877524521079
// MI455X (gfx1250) — compile-verified, matrix operand order checked by reading, not run on hardware
//
#include <hip/hip_runtime.h>
#include <hip/hip_bf16.h>
#include <cstdint>

// ---------------- problem constants (match reference) ----------------
#define BB    4
#define NN    50000
#define TOTP  (BB * NN)       // 200000 points
#define CF    32
#define GRES  32
#define PRES  128
#define HID   256
#define NBLK  5
#define CDIM  128             // 4*CF
#define ROWS  32              // points per workgroup (2 WMMA m-tiles)

typedef float  v8f   __attribute__((ext_vector_type(8)));
typedef __bf16 v8bf  __attribute__((ext_vector_type(8)));
typedef __bf16 v16bf __attribute__((ext_vector_type(16)));

// ---------------------------------------------------------------------
// Kernel 1: fp32 weights -> bf16, transposed to [n][k] so WMMA B-frags
// (per-lane: 8 bf16 @ k0, 8 bf16 @ k0+16) are two contiguous 16B loads.
// Total bf16 weights = 1.6MB -> L2 resident for the whole dispatch.
// ---------------------------------------------------------------------
__global__ __launch_bounds__(256) void prep_kernel(
    const float* __restrict__ Wc, const float* __restrict__ W0,
    const float* __restrict__ W1,
    __bf16* __restrict__ WcT, __bf16* __restrict__ W0T, __bf16* __restrict__ W1T) {
  int i = blockIdx.x * blockDim.x + threadIdx.x;
  if (i < NBLK * CDIM * HID) {              // Wc: (blk,k,n) -> (blk,n,k)
    int n   = i % HID;
    int k   = (i / HID) % CDIM;
    int blk = i / (HID * CDIM);
    WcT[((size_t)blk * HID + n) * CDIM + k] = (__bf16)Wc[i];
  }
  if (i < NBLK * HID * HID) {               // W0/W1: (blk,k,n) -> (blk,n,k)
    int n   = i % HID;
    int k   = (i / HID) % HID;
    int blk = i / (HID * HID);
    size_t o = ((size_t)blk * HID + n) * HID + k;
    W0T[o] = (__bf16)W0[i];
    W1T[o] = (__bf16)W1[i];
  }
}

// ---------------------------------------------------------------------
// Kernel 2: feature sampling. thread = (point, channel); lane == channel
// group so each wave32 samples exactly one source (no divergence).
// Output: Cf[point][128] bf16 (row-major, WMMA-A friendly).
// ---------------------------------------------------------------------
__global__ __launch_bounds__(256) void sample_kernel(
    const float* __restrict__ p,  const float* __restrict__ cg,
    const float* __restrict__ cxy, const float* __restrict__ cyz,
    const float* __restrict__ cxz, __bf16* __restrict__ Cf) {
  long tid = (long)blockIdx.x * blockDim.x + threadIdx.x;
  if (tid >= (long)TOTP * 128) return;
  int c  = (int)(tid & 127);
  int pi = (int)(tid >> 7);
  int b  = pi / NN;

  float px = p[pi * 3 + 0], py = p[pi * 3 + 1], pz = p[pi * 3 + 2];
  const float inv = 1.0f / 1.101f;                 // 1/(1+PAD+1e-3)
  float xn = fminf(fmaxf(px * inv + 0.5f, 0.0f), 1.0f - 1e-5f);
  float yn = fminf(fmaxf(py * inv + 0.5f, 0.0f), 1.0f - 1e-5f);
  float zn = fminf(fmaxf(pz * inv + 0.5f, 0.0f), 1.0f - 1e-5f);
  // (g+1)*0.5 == xn exactly, and xn*(W-1) is already in-range.

  int group = c >> 5, ch = c & 31;
  float val;
  if (group == 0) {                                // trilinear from grid
    float ix = xn * (GRES - 1), iy = yn * (GRES - 1), iz = zn * (GRES - 1);
    int   x0 = (int)ix, y0 = (int)iy, z0 = (int)iz;
    float wx = ix - x0, wy = iy - y0, wz = iz - z0;
    int   x1 = min(x0 + 1, GRES - 1), y1 = min(y0 + 1, GRES - 1), z1 = min(z0 + 1, GRES - 1);
    const float* g = cg + ((size_t)b * CF + ch) * (GRES * GRES * GRES);
    auto at = [&](int z, int y, int x) { return g[((size_t)z * GRES + y) * GRES + x]; };
    val = at(z0,y0,x0)*(1.0f-wx)*(1.0f-wy)*(1.0f-wz) + at(z0,y0,x1)*wx*(1.0f-wy)*(1.0f-wz)
        + at(z0,y1,x0)*(1.0f-wx)*wy*(1.0f-wz)        + at(z0,y1,x1)*wx*wy*(1.0f-wz)
        + at(z1,y0,x0)*(1.0f-wx)*(1.0f-wy)*wz        + at(z1,y0,x1)*wx*(1.0f-wy)*wz
        + at(z1,y1,x0)*(1.0f-wx)*wy*wz               + at(z1,y1,x1)*wx*wy*wz;
  } else {                                         // bilinear from a plane
    float u, v; const float* pl;
    if (group == 1)      { u = xn; v = yn; pl = cxy; }
    else if (group == 2) { u = yn; v = zn; pl = cyz; }
    else                 { u = xn; v = zn; pl = cxz; }
    float ix = u * (PRES - 1), iy = v * (PRES - 1);
    int   x0 = (int)ix, y0 = (int)iy;
    float wx = ix - x0, wy = iy - y0;
    int   x1 = min(x0 + 1, PRES - 1), y1 = min(y0 + 1, PRES - 1);
    const float* g = pl + ((size_t)b * CF + ch) * (PRES * PRES);
    val = g[(size_t)y0 * PRES + x0] * (1.0f-wx) * (1.0f-wy)
        + g[(size_t)y0 * PRES + x1] * wx        * (1.0f-wy)
        + g[(size_t)y1 * PRES + x0] * (1.0f-wx) * wy
        + g[(size_t)y1 * PRES + x1] * wx        * wy;
  }
  Cf[tid] = (__bf16)val;
}

// ---------------------------------------------------------------------
// fragment loader: ISA 16-bit A/B layout -> per lane: row/col = lane%16,
// 8 bf16 at k0=(lane/16)*8 and 8 bf16 at k0+16 (both 16B aligned).
// ---------------------------------------------------------------------
__device__ __forceinline__ v16bf frag16(const __bf16* ptr) {
  v8bf lo = *(const v8bf*)ptr;
  v8bf hi = *(const v8bf*)(ptr + 16);
  v16bf r;
#pragma unroll
  for (int i = 0; i < 8; ++i) { r[i] = lo[i]; r[i + 8] = hi[i]; }
  return r;
}

// ---------------------------------------------------------------------
// Kernel 3: ResNet MLP. 4 wave32 / block, 32 points / block (MT=2
// m-tiles): every 1KB B-fragment feeds 2 independent WMMAs (32 FLOP per
// L2 byte). Per k-step, all 4 B-fragments are fetched into distinct
// registers first so 8 global_load_b128 stay in flight and the waits
// can retire with slack instead of draining to zero before each WMMA.
// Wave w owns hidden columns [64w, 64w+64) as 4 n-tiles.
// ---------------------------------------------------------------------
__global__ __launch_bounds__(128) void mlp_kernel(
    const __bf16* __restrict__ Cf,
    const __bf16* __restrict__ WcT, const __bf16* __restrict__ W0T,
    const __bf16* __restrict__ W1T,
    const float* __restrict__ p,
    const float* __restrict__ Wp,  const float* __restrict__ bp,
    const float* __restrict__ bc,  const float* __restrict__ b0,
    const float* __restrict__ b1,
    const float* __restrict__ Wout, const float* __restrict__ bout,
    float* __restrict__ out) {
  __shared__ __bf16 Ac[ROWS][CDIM + 8];   // point features (A of c@Wc)
  __shared__ __bf16 Xs[ROWS][HID + 8];    // relu(net) bf16
  __shared__ __bf16 Ys[ROWS][HID + 8];    // relu(h) bf16
  __shared__ float  Ps[ROWS][4];          // raw coords
  __shared__ float  Outs[ROWS];

  const int tid  = threadIdx.x;
  const int lane = tid & 31;
  const int wave = tid >> 5;
  const int lo   = lane & 15;             // row-in-tile (A) / col (B) / N (C)
  const int hiH  = lane >> 4;             // K-half / M-half select
  const int kh   = hiH * 8;
  const int nbase = wave * 64;
  const int m0   = blockIdx.x * ROWS;

  // stage the 32x128 feature tile into LDS (each thread: 32 bf16)
  {
    int r = tid >> 2, seg = tid & 3;      // 128 threads -> 32 rows x 4 segs
    int pi = m0 + r;
    v8bf z;
#pragma unroll
    for (int i = 0; i < 8; ++i) z[i] = (__bf16)0.0f;
    v8bf a0 = z, a1 = z, a2 = z, a3 = z;
    if (pi < TOTP) {
      const v8bf* s = (const v8bf*)(Cf + (size_t)pi * CDIM + seg * 32);
      a0 = s[0]; a1 = s[1]; a2 = s[2]; a3 = s[3];
    }
    v8bf* d = (v8bf*)&Ac[r][seg * 32];
    d[0] = a0; d[1] = a1; d[2] = a2; d[3] = a3;
  }
  if (tid < ROWS) {
    int pi = m0 + tid;
#pragma unroll
    for (int j = 0; j < 3; ++j) Ps[tid][j] = (pi < TOTP) ? p[pi * 3 + j] : 0.0f;
    Outs[tid] = 0.0f;
  }
  __syncthreads();

  // net = p @ Wp + bp, directly in WMMA C-layout (VALU, tiny K=3)
  v8f acc[2][4];
#pragma unroll
  for (int t = 0; t < 4; ++t) {
    int n = nbase + t * 16 + lo;
    float w0n = Wp[n], w1n = Wp[HID + n], w2n = Wp[2 * HID + n], bpn = bp[n];
#pragma unroll
    for (int mt = 0; mt < 2; ++mt) {
#pragma unroll
      for (int v = 0; v < 8; ++v) {
        int m = mt * 16 + v + 8 * hiH;
        acc[mt][t][v] = Ps[m][0] * w0n + Ps[m][1] * w1n + Ps[m][2] * w2n + bpn;
      }
    }
  }

#pragma unroll 1
  for (int blk = 0; blk < NBLK; ++blk) {
    // ---- net += c @ Wc[blk] + bc[blk]  (K = 128) ----
#pragma unroll
    for (int kt = 0; kt < 4; ++kt) {
      v16bf a0 = frag16(&Ac[lo][kt * 32 + kh]);
      v16bf a1 = frag16(&Ac[16 + lo][kt * 32 + kh]);
      v16bf bfr[4];
#pragma unroll
      for (int t = 0; t < 4; ++t)
        bfr[t] = frag16(WcT + ((size_t)(blk * HID + nbase + t * 16 + lo)) * CDIM + kt * 32 + kh);
#pragma unroll
      for (int t = 0; t < 4; ++t) {
        acc[0][t] = __builtin_amdgcn_wmma_f32_16x16x32_bf16(
            false, a0, false, bfr[t], (short)0, acc[0][t], false, false);
        acc[1][t] = __builtin_amdgcn_wmma_f32_16x16x32_bf16(
            false, a1, false, bfr[t], (short)0, acc[1][t], false, false);
      }
    }
#pragma unroll
    for (int t = 0; t < 4; ++t) {
      float bb = bc[blk * HID + nbase + t * 16 + lo];
#pragma unroll
      for (int mt = 0; mt < 2; ++mt)
#pragma unroll
        for (int v = 0; v < 8; ++v) acc[mt][t][v] += bb;
    }

    __syncthreads();                       // prior iter's Xs/Ys reads done
    // ---- Xs = relu(net) (bf16) ----
#pragma unroll
    for (int t = 0; t < 4; ++t) {
      int n = nbase + t * 16 + lo;
#pragma unroll
      for (int mt = 0; mt < 2; ++mt)
#pragma unroll
        for (int v = 0; v < 8; ++v) {
          float x = acc[mt][t][v];
          Xs[mt * 16 + v + 8 * hiH][n] = (__bf16)(x > 0.0f ? x : 0.0f);
        }
    }
    __syncthreads();

    // ---- h = relu(net) @ W0[blk] + b0[blk]  (K = 256) ----
    v8f hacc[2][4];
#pragma unroll
    for (int t = 0; t < 4; ++t) {
      float bb = b0[blk * HID + nbase + t * 16 + lo];
#pragma unroll
      for (int mt = 0; mt < 2; ++mt)
#pragma unroll
        for (int v = 0; v < 8; ++v) hacc[mt][t][v] = bb;
    }
#pragma unroll
    for (int kt = 0; kt < 8; ++kt) {
      v16bf a0 = frag16(&Xs[lo][kt * 32 + kh]);
      v16bf a1 = frag16(&Xs[16 + lo][kt * 32 + kh]);
      v16bf bfr[4];
#pragma unroll
      for (int t = 0; t < 4; ++t)
        bfr[t] = frag16(W0T + ((size_t)(blk * HID + nbase + t * 16 + lo)) * HID + kt * 32 + kh);
#pragma unroll
      for (int t = 0; t < 4; ++t) {
        hacc[0][t] = __builtin_amdgcn_wmma_f32_16x16x32_bf16(
            false, a0, false, bfr[t], (short)0, hacc[0][t], false, false);
        hacc[1][t] = __builtin_amdgcn_wmma_f32_16x16x32_bf16(
            false, a1, false, bfr[t], (short)0, hacc[1][t], false, false);
      }
    }
    // ---- Ys = relu(h) (bf16) ----
#pragma unroll
    for (int t = 0; t < 4; ++t) {
      int n = nbase + t * 16 + lo;
#pragma unroll
      for (int mt = 0; mt < 2; ++mt)
#pragma unroll
        for (int v = 0; v < 8; ++v) {
          float x = hacc[mt][t][v];
          Ys[mt * 16 + v + 8 * hiH][n] = (__bf16)(x > 0.0f ? x : 0.0f);
        }
    }
    __syncthreads();

    // ---- net += relu(h) @ W1[blk] + b1[blk]  (K = 256) ----
#pragma unroll
    for (int kt = 0; kt < 8; ++kt) {
      v16bf a0 = frag16(&Ys[lo][kt * 32 + kh]);
      v16bf a1 = frag16(&Ys[16 + lo][kt * 32 + kh]);
      v16bf bfr[4];
#pragma unroll
      for (int t = 0; t < 4; ++t)
        bfr[t] = frag16(W1T + ((size_t)(blk * HID + nbase + t * 16 + lo)) * HID + kt * 32 + kh);
#pragma unroll
      for (int t = 0; t < 4; ++t) {
        acc[0][t] = __builtin_amdgcn_wmma_f32_16x16x32_bf16(
            false, a0, false, bfr[t], (short)0, acc[0][t], false, false);
        acc[1][t] = __builtin_amdgcn_wmma_f32_16x16x32_bf16(
            false, a1, false, bfr[t], (short)0, acc[1][t], false, false);
      }
    }
#pragma unroll
    for (int t = 0; t < 4; ++t) {
      float bb = b1[blk * HID + nbase + t * 16 + lo];
#pragma unroll
      for (int mt = 0; mt < 2; ++mt)
#pragma unroll
        for (int v = 0; v < 8; ++v) acc[mt][t][v] += bb;
    }
  }

  // ---- out = relu(net) @ Wout + bout (rank-1: VALU + LDS atomics) ----
  float part[2][8];
#pragma unroll
  for (int mt = 0; mt < 2; ++mt)
#pragma unroll
    for (int v = 0; v < 8; ++v) part[mt][v] = 0.0f;
#pragma unroll
  for (int t = 0; t < 4; ++t) {
    float wn = Wout[nbase + t * 16 + lo];
#pragma unroll
    for (int mt = 0; mt < 2; ++mt)
#pragma unroll
      for (int v = 0; v < 8; ++v) {
        float x = acc[mt][t][v];
        part[mt][v] += (x > 0.0f ? x : 0.0f) * wn;
      }
  }
#pragma unroll
  for (int mt = 0; mt < 2; ++mt)
#pragma unroll
    for (int v = 0; v < 8; ++v)
      atomicAdd(&Outs[mt * 16 + v + 8 * hiH], part[mt][v]);
  __syncthreads();
  if (tid < ROWS) {
    int pi = m0 + tid;
    if (pi < TOTP) out[pi] = Outs[tid] + bout[0];
  }
}

// ---------------------------------------------------------------------
extern "C" void kernel_launch(void* const* d_in, const int* in_sizes, int n_in,
                              void* d_out, int out_size, void* d_ws, size_t ws_size,
                              hipStream_t stream) {
  (void)in_sizes; (void)n_in; (void)out_size; (void)ws_size;
  const float* p    = (const float*)d_in[0];
  const float* cg   = (const float*)d_in[1];
  const float* cxy  = (const float*)d_in[2];
  const float* cyz  = (const float*)d_in[3];
  const float* cxz  = (const float*)d_in[4];
  const float* Wp   = (const float*)d_in[5];
  const float* bp   = (const float*)d_in[6];
  const float* Wc   = (const float*)d_in[7];
  const float* bc   = (const float*)d_in[8];
  const float* W0   = (const float*)d_in[9];
  const float* b0   = (const float*)d_in[10];
  const float* W1   = (const float*)d_in[11];
  const float* b1   = (const float*)d_in[12];
  const float* Wout = (const float*)d_in[13];
  const float* bout = (const float*)d_in[14];
  float* out = (float*)d_out;

  // workspace layout (bf16 elements): WcT | W0T | W1T | Cfeat  (~52.8 MB)
  __bf16* WcT = (__bf16*)d_ws;
  __bf16* W0T = WcT + (size_t)NBLK * HID * CDIM;
  __bf16* W1T = W0T + (size_t)NBLK * HID * HID;
  __bf16* Cf  = W1T + (size_t)NBLK * HID * HID;

  {
    int tot = NBLK * HID * HID;                       // covers Wc range too
    prep_kernel<<<(tot + 255) / 256, 256, 0, stream>>>(Wc, W0, W1, WcT, W0T, W1T);
  }
  {
    long tot = (long)TOTP * 128;
    sample_kernel<<<(int)((tot + 255) / 256), 256, 0, stream>>>(p, cg, cxy, cyz, cxz, Cf);
  }
  {
    int nb = (TOTP + ROWS - 1) / ROWS;                // 6250 blocks, 4 waves each
    mlp_kernel<<<nb, 128, 0, stream>>>(Cf, WcT, W0T, W1T, p, Wp, bp, bc, b0, b1,
                                       Wout, bout, out);
  }
}
